// Attention1_62921270886468
// MI455X (gfx1250) — matrix operand order repaired and checked by reading, not compiled, not run on hardware
//
#include <hip/hip_runtime.h>
#include <hip/hip_bf16.h>
#include <math.h>

// Problem constants (reference: B=8, L=2048, D=256)
#define B_  8
#define L_  2048
#define D_  256

typedef __attribute__((ext_vector_type(16))) __bf16  v16bf;
typedef __attribute__((ext_vector_type(8)))  float   v8f;
typedef __attribute__((__vector_size__(16))) int     vsi4;   // matches async-LDS builtin param
typedef __attribute__((ext_vector_type(4))) unsigned int u32x4;
typedef __attribute__((ext_vector_type(8))) int          i32x8;
typedef __attribute__((ext_vector_type(4))) int          i32x4;

// ---- feature probes (compile-safe: fall back if builtin not declared) ----
#if defined(__gfx1250__) && __has_builtin(__builtin_amdgcn_global_load_async_to_lds_b128) && \
    __has_builtin(__builtin_amdgcn_s_wait_asynccnt)
#define USE_ASYNC_LDS 1
#else
#define USE_ASYNC_LDS 0
#endif

#if defined(__gfx1250__) && __has_builtin(__builtin_amdgcn_tensor_load_to_lds) && \
    __has_builtin(__builtin_amdgcn_s_wait_tensorcnt)
#define USE_TDM 1
#else
#define USE_TDM 0
#endif

// Convert a pair of f32 to bf16 (RTE). Prefer the packed HW convert.
__device__ __forceinline__ void cvt2(__bf16* dst, float a, float b) {
#if __has_builtin(__builtin_amdgcn_cvt_pk_bf16_f32)
  auto r = __builtin_amdgcn_cvt_pk_bf16_f32(a, b);
  union { decltype(r) rv; __bf16 h[2]; } u;
  u.rv = r;
  dst[0] = u.h[0]; dst[1] = u.h[1];
#else
  dst[0] = (__bf16)a; dst[1] = (__bf16)b;   // native fptrunc float->bf16
#endif
}

union AFrag {            // one wave-striped 16x32 bf16 WMMA operand (8 VGPRs/lane)
  v16bf v;
  __bf16 h[16];
};

__device__ __forceinline__ void pack4(__bf16* u, float4 f) {
  cvt2(u + 0, f.x, f.y);
  cvt2(u + 2, f.z, f.w);
}

// ---------------------------------------------------------------------------
// Kernel 1: q = x @ W^T + b  (grid.y==0 -> Q from query/Wq, ==1 -> K from key/Wk)
// One 16-row output tile per block; 8 waves each own 2 of the 16 column tiles.
// ---------------------------------------------------------------------------
__global__ void proj_kernel(const float* __restrict__ query,
                            const float* __restrict__ key_,
                            const float* __restrict__ Wq_w, const float* __restrict__ Wq_b,
                            const float* __restrict__ Wk_w, const float* __restrict__ Wk_b,
                            __bf16* __restrict__ Qb,
                            __bf16* __restrict__ Kb)
{
  const int lane = threadIdx.x & 31;
  const int w    = threadIdx.x >> 5;          // wave 0..7
  const int row0 = blockIdx.x * 16;
  const bool isK = (blockIdx.y != 0);

  const float* src = isK ? key_ : query;
  const float* Ww  = isK ? Wk_w : Wq_w;
  const float* Wb  = isK ? Wk_b : Wq_b;
  __bf16* dst = isK ? Kb : Qb;

  const int nlo  = lane & 15;
  const int hi   = lane >> 4;                 // half-wave select
  const int kb8  = hi * 8;                    // A-fragment K base
  const int kb16 = hi * 16;                   // unused here; B fragment also uses kb8

  // A fragments (16 rows x 32 k) for all 8 k-chunks of D=256; reused by both col tiles.
  AFrag a[8];
  {
    const float* arow = src + (size_t)(row0 + nlo) * D_;
#pragma unroll
    for (int c = 0; c < 8; ++c) {
      const float* p = arow + c * 32 + kb8;
      pack4(&a[c].h[0],  *(const float4*)(p + 0));    // K = kb8+0..3   -> e0..3
      pack4(&a[c].h[4],  *(const float4*)(p + 4));    // K = kb8+4..7   -> e4..7
      pack4(&a[c].h[8],  *(const float4*)(p + 16));   // K = kb8+16..19 -> e8..11
      pack4(&a[c].h[12], *(const float4*)(p + 20));   // K = kb8+20..23 -> e12..15
    }
  }

#pragma unroll
  for (int tt = 0; tt < 2; ++tt) {
    const int col0 = (w + tt * 8) * 16;
    v8f acc;
    const float bias = Wb[col0 + nlo];
#pragma unroll
    for (int i = 0; i < 8; ++i) acc[i] = bias;

#pragma unroll
    for (int c = 0; c < 8; ++c) {
      // B[k,n] = W[col0+n, d0+k] : one W row, contiguous in d.
      const float* p = Ww + (size_t)(col0 + nlo) * D_ + c * 32 + kb8;
      AFrag bf;
      pack4(&bf.h[0],  *(const float4*)(p + 0));
      pack4(&bf.h[4],  *(const float4*)(p + 4));
      pack4(&bf.h[8],  *(const float4*)(p + 16));
      pack4(&bf.h[12], *(const float4*)(p + 20));
      acc = __builtin_amdgcn_wmma_f32_16x16x32_bf16(false, a[c].v, false, bf.v,
                                                    (short)0, acc, false, false);
    }

#pragma unroll
    for (int i = 0; i < 8; ++i) {
      const int r = row0 + i + 8 * hi;        // C layout: VGPR i -> row i + 8*hi
      dst[(size_t)r * D_ + col0 + nlo] = (__bf16)acc[i];
    }
  }
}

// ---------------------------------------------------------------------------
// Kernel 2: V fp32 -> bf16
// ---------------------------------------------------------------------------
__global__ void cvt_v_kernel(const float* __restrict__ v,
                             __bf16* __restrict__ vb, int n4)
{
  int i = blockIdx.x * blockDim.x + threadIdx.x;
  if (i < n4) {
    float4 f = ((const float4*)v)[i];
    union { __bf16 h[4]; uint2 u; } o;
    cvt2(&o.h[0], f.x, f.y);
    cvt2(&o.h[2], f.z, f.w);
    ((uint2*)vb)[i] = o.u;
  }
}

// ---------------------------------------------------------------------------
// Kernel 3: flash attention. Block = 4 waves; each wave owns 16 query rows.
// K tile: async global->LDS DMA (ASYNCcnt).  V tile: TDM tensor_load_to_lds
// (TENSORcnt) row-major, then a cheap LDS->LDS transpose for the PV operand.
// ---------------------------------------------------------------------------
__global__ __launch_bounds__(128)
void flash_kernel(const __bf16* __restrict__ Qb,
                  const __bf16* __restrict__ Kb,
                  const __bf16* __restrict__ Vb,
                  const int* __restrict__ mask,
                  float* __restrict__ out)
{
  __shared__ __bf16 sK[32 * D_];       // [key][d]
  __shared__ __bf16 sVt[D_ * 32];      // [d][key]  (transposed, PV B-operand)
#if USE_TDM
  __shared__ __bf16 sV[32 * D_];       // [key][d]  row-major TDM landing buffer
#endif
  __shared__ __bf16 sP[4][16 * 32];    // per-wave P staging (C->A transpose)

  const int tid  = threadIdx.x;
  const int lane = tid & 31;
  const int w    = tid >> 5;
  const int b    = blockIdx.y;
  const int q0   = blockIdx.x * 64 + w * 16;

  const int nlo  = lane & 15;
  const int hi   = lane >> 4;
  const int kb8  = hi * 8;
  const int kb16 = hi * 16;

  // Q A-fragments for all 8 k-chunks, resident across the whole key loop.
  AFrag qa[8];
  {
    const __bf16* qrow = Qb + ((size_t)b * L_ + q0 + nlo) * D_;
#pragma unroll
    for (int c = 0; c < 8; ++c) {
      const __bf16* p = qrow + c * 32 + kb16;
      *(uint4*)&qa[c].h[0] = *(const uint4*)(p);        // e0..7  <- K kb16+0..7
      *(uint4*)&qa[c].h[8] = *(const uint4*)(p + 8);    // e8..15 <- K kb16+8..15
    }
  }

  v8f o[16];
#pragma unroll
  for (int t = 0; t < 16; ++t)
#pragma unroll
    for (int i = 0; i < 8; ++i) o[t][i] = 0.f;

  float mrow[8], lrow[8];
#pragma unroll
  for (int i = 0; i < 8; ++i) { mrow[i] = -3.0e38f; lrow[i] = 0.f; }

  const int* mptr = mask + b * L_;
  const float scale = 0.0625f;                 // 1/sqrt(256)

  for (int kt = 0; kt < L_ / 32; ++kt) {
    const int k0 = kt * 32;

    // ---- stage K tile (async DMA) ----
    {
      const uint4* gK = (const uint4*)(Kb + ((size_t)b * L_ + k0) * D_);
      uint4* sKv = (uint4*)sK;
#if USE_ASYNC_LDS
#pragma unroll
      for (int i = 0; i < 8; ++i) {
        const int idx = tid + i * 128;
        __builtin_amdgcn_global_load_async_to_lds_b128(
            (vsi4*)(gK + idx),           // src: global, 16B per lane
            (vsi4*)(sKv + idx),          // dst: LDS
            0, 0);
      }
#else
#pragma unroll
      for (int i = 0; i < 8; ++i)
        sKv[tid + i * 128] = gK[tid + i * 128];
#endif
      if (kt + 1 < L_ / 32)                    // global_prefetch_b8 for next K tile
        __builtin_prefetch(Kb + ((size_t)b * L_ + k0 + 32) * D_ + tid * 16, 0, 1);
    }

    // ---- stage V tile ----
#if USE_TDM
    if (w == 0) {
      // Tensor DMA: 32x256 bf16 2-D tile, row stride 256 elements.
      const unsigned long long gaddr =
          (unsigned long long)(size_t)(const void*)(Vb + ((size_t)b * L_ + k0) * D_);
      u32x4 g0;
      g0[0] = 1u;                                          // count=1 valid descriptor
      g0[1] = (unsigned)(size_t)(void*)sV;                 // lds_addr (bytes)
      g0[2] = (unsigned)(gaddr & 0xFFFFFFFFu);             // global_addr[31:0]
      g0[3] = (unsigned)((gaddr >> 32) & 0x1FFFFFFu)       // global_addr[56:32]
            | (2u << 30);                                  // type = 2 ("image")
      i32x8 g1;
      g1[0] = 1 << 16;                                     // data_size=1 (2-byte elems)
      g1[1] = (int)(256u << 16);                           // tensor_dim0 = 256 (bits 63:48)
      g1[2] = (int)(32u  << 16);                           // tensor_dim1 = 32  (bits 95:80)
      g1[3] = (int)(256u << 16);                           // tile_dim0 = 256   (bits 127:112)
      g1[4] = 32;                                          // tile_dim1 = 32, tile_dim2 = 0
      g1[5] = 256;                                         // tensor_dim0_stride = 256
      g1[6] = 0;
      g1[7] = 0;
      i32x4 gz4 = {0, 0, 0, 0};                            // groups 2/3 unused (2-D tensor)
      i32x8 gz8 = {0, 0, 0, 0, 0, 0, 0, 0};
      __builtin_amdgcn_tensor_load_to_lds(g0, g1, gz4, gz4, gz8, 0);
      __builtin_amdgcn_s_wait_tensorcnt(0);                // wave 0 drains TENSORcnt
    }
#else
    {
      const __bf16* gV = Vb + ((size_t)b * L_ + k0) * D_;
      for (int idx = tid; idx < 32 * D_; idx += 128) {
        const int key = idx >> 8;              // idx = key*256 + d
        const int d   = idx & 255;
        sVt[d * 32 + key] = gV[idx];
      }
    }
#endif

#if USE_ASYNC_LDS
    __builtin_amdgcn_s_wait_asynccnt(0);       // drain this wave's async DMAs
#endif
    __syncthreads();                           // K (and TDM'd V) visible to all waves

#if USE_TDM
    // LDS->LDS transpose of the DMA'd tile: sV[key][d] -> sVt[d][key].
    for (int idx = tid; idx < 32 * D_; idx += 128) {
      const int key = idx >> 8;
      const int d   = idx & 255;
      sVt[d * 32 + key] = sV[idx];
    }
    // sVt is consumed only after the mid-iteration barrier below.
#endif

    // ---- S = Q K^T : two 16x16 subtiles, 8 bf16 WMMAs each ----
    v8f s0, s1;
#pragma unroll
    for (int i = 0; i < 8; ++i) { s0[i] = 0.f; s1[i] = 0.f; }
#pragma unroll
    for (int c = 0; c < 8; ++c) {
      const __bf16* pk0 = sK + (size_t)nlo * D_ + c * 32 + kb16;
      AFrag bk0;                               // B[k,n] = K[n, d0+k]
      *(uint4*)&bk0.h[0] = *(const uint4*)(pk0);
      *(uint4*)&bk0.h[8] = *(const uint4*)(pk0 + 8);
      s0 = __builtin_amdgcn_wmma_f32_16x16x32_bf16(false, qa[c].v, false, bk0.v,
                                                   (short)0, s0, false, false);
      const __bf16* pk1 = sK + (size_t)(16 + nlo) * D_ + c * 32 + kb16;
      AFrag bk1;
      *(uint4*)&bk1.h[0] = *(const uint4*)(pk1);
      *(uint4*)&bk1.h[8] = *(const uint4*)(pk1 + 8);
      s1 = __builtin_amdgcn_wmma_f32_16x16x32_bf16(false, qa[c].v, false, bk1.v,
                                                   (short)0, s1, false, false);
    }

    // ---- mask (mult+additive like the reference), online softmax ----
    const int m0 = mptr[k0 + nlo];
    const int m1 = mptr[k0 + 16 + nlo];
#pragma unroll
    for (int i = 0; i < 8; ++i) {
      s0[i] = m0 ? s0[i] * scale : -1.0e9f;
      s1[i] = m1 ? s1[i] * scale : -1.0e9f;
    }
#pragma unroll
    for (int i = 0; i < 8; ++i) {
      // row i+8*hi lives across one 16-lane group -> xor-reduce within width 16
      float v = fmaxf(s0[i], s1[i]);
      v = fmaxf(v, __shfl_xor(v, 1, 16));
      v = fmaxf(v, __shfl_xor(v, 2, 16));
      v = fmaxf(v, __shfl_xor(v, 4, 16));
      v = fmaxf(v, __shfl_xor(v, 8, 16));
      const float mnew = fmaxf(mrow[i], v);
      const float corr = __expf(mrow[i] - mnew);
      const float p0 = __expf(s0[i] - mnew);
      const float p1 = __expf(s1[i] - mnew);
      float rs = p0 + p1;
      rs += __shfl_xor(rs, 1, 16);
      rs += __shfl_xor(rs, 2, 16);
      rs += __shfl_xor(rs, 4, 16);
      rs += __shfl_xor(rs, 8, 16);
      lrow[i] = lrow[i] * corr + rs;
      mrow[i] = mnew;
      s0[i] = p0; s1[i] = p1;
#pragma unroll
      for (int t = 0; t < 16; ++t) o[t][i] *= corr;
    }

    // ---- P: C-layout -> A-layout via per-wave LDS staging ----
    __bf16* pbuf = &sP[w][0];
#pragma unroll
    for (int i = 0; i < 8; ++i) {
      const int r = i + 8 * hi;
      pbuf[r * 32 + nlo]      = (__bf16)s0[i];
      pbuf[r * 32 + 16 + nlo] = (__bf16)s1[i];
    }
    __syncthreads();                           // publishes P and sVt transposes

    AFrag pa;
    {
      const __bf16* pp = pbuf + nlo * 32 + kb8;
      *(uint4*)&pa.h[0] = *(const uint4*)(pp);        // e0..7  <- key kb8+0..7
      *(uint4*)&pa.h[8] = *(const uint4*)(pp + 16);   // e8..15 <- key kb8+16..23
    }

    // ---- O += P V : 16 bf16 WMMAs across D=256 ----
#pragma unroll
    for (int t = 0; t < 16; ++t) {
      const __bf16* pv = sVt + (size_t)(t * 16 + nlo) * 32 + kb8;
      AFrag bv;                                // B[k,n] = V[key=k, d=n], contiguous via sVt
      *(uint4*)&bv.h[0] = *(const uint4*)(pv);
      *(uint4*)&bv.h[8] = *(const uint4*)(pv + 16);
      o[t] = __builtin_amdgcn_wmma_f32_16x16x32_bf16(false, pa.v, false, bv.v,
                                                     (short)0, o[t], false, false);
    }
    __syncthreads();
  }

  // ---- normalize by softmax denominator and store fp32 output ----
#pragma unroll
  for (int t = 0; t < 16; ++t) {
#pragma unroll
    for (int i = 0; i < 8; ++i) {
      const int r = q0 + i + 8 * hi;
      out[((size_t)b * L_ + r) * D_ + t * 16 + nlo] = o[t][i] / lrow[i];
    }
  }
}

// ---------------------------------------------------------------------------
// Launch: inputs in setup_inputs() order. Workspace: bf16 Q, K, V (3 x 8 MB).
// ---------------------------------------------------------------------------
extern "C" void kernel_launch(void* const* d_in, const int* in_sizes, int n_in,
                              void* d_out, int out_size, void* d_ws, size_t ws_size,
                              hipStream_t stream)
{
  const float* query = (const float*)d_in[0];
  const float* key_  = (const float*)d_in[1];
  const float* value = (const float*)d_in[2];
  const int*   mask  = (const int*)d_in[3];
  const float* Wq_w  = (const float*)d_in[4];
  const float* Wq_b  = (const float*)d_in[5];
  const float* Wk_w  = (const float*)d_in[6];
  const float* Wk_b  = (const float*)d_in[7];
  float* out = (float*)d_out;

  const size_t nBLD = (size_t)B_ * L_ * D_;          // 4,194,304 elements
  __bf16* Qb = (__bf16*)d_ws;                        // 8 MB
  __bf16* Kb = Qb + nBLD;                            // 8 MB
  __bf16* Vb = Kb + nBLD;                            // 8 MB (24 MB total)

  dim3 gA(B_ * L_ / 16, 2);
  proj_kernel<<<gA, 256, 0, stream>>>(query, key_, Wq_w, Wq_b, Wk_w, Wk_b, Qb, Kb);

  const int n4 = (int)(nBLD / 4);
  cvt_v_kernel<<<(n4 + 255) / 256, 256, 0, stream>>>(value, Vb, n4);

  dim3 gB(L_ / 64, B_);
  flash_kernel<<<gB, 128, 0, stream>>>(Qb, Kb, Vb, mask, out);
}
